// ResMambaGenerator_67980742361142
// MI455X (gfx1250) — matrix operand order repaired and checked by reading, not compiled, not run on hardware
//
#include <hip/hip_runtime.h>
#include <hip/hip_bf16.h>

typedef __bf16 bf16;
typedef __attribute__((ext_vector_type(16))) __bf16 v16bf;
typedef __attribute__((ext_vector_type(8)))  __bf16 bf16x8;
typedef __attribute__((ext_vector_type(8)))  float  v8f;

#define EPSN 1e-5f
#define LDS_STRIDE 40   // bf16 units per B-tile column (pad 32 -> 40: conflict-free b128)

union V16 { v16bf v; bf16x8 h[2]; };

__device__ __forceinline__ float act_f(float x, int act) {
  if (act == 1) return fmaxf(x, 0.f);                          // relu
  if (act == 2) return (x > 20.f) ? x : log1pf(__expf(x));     // softplus
  if (act == 3) return tanhf(x);                               // tanh
  return x;
}

__device__ __forceinline__ int iclampi(int v, int hi) {
  v = v < 0 ? 0 : v;
  return v > hi ? hi : v;
}

// ---------------- packing ----------------
__global__ void pack_bf16_kernel(const float* __restrict__ src, bf16* __restrict__ dst, int n) {
  int i = blockIdx.x * blockDim.x + threadIdx.x;
  if (i < n) dst[i] = (bf16)src[i];
}

// rows x K (f32) -> rows x Kp (bf16), zero-padded K tail
__global__ void pack_wpad_kernel(const float* __restrict__ src, bf16* __restrict__ dst,
                                 int K, int Kp, long long total) {
  long long i = (long long)blockIdx.x * blockDim.x + threadIdx.x;
  if (i >= total) return;
  int k = (int)(i % Kp);
  long long r = i / Kp;
  dst[i] = (k < K) ? (bf16)src[(size_t)r * K + k] : (bf16)0.f;
}

// W: [Cin][Cout][3][3] -> dst: [Cout][Cin*9] spatially flipped (transpose conv)
__global__ void pack_convT_kernel(const float* __restrict__ W, bf16* __restrict__ dst,
                                  int Cin, int Cout) {
  int i = blockIdx.x * blockDim.x + threadIdx.x;
  int total = Cin * Cout * 9;
  if (i >= total) return;
  int co = i / (Cin * 9);
  int r  = i % (Cin * 9);
  int ci = r / 9;
  int rr = r % 9;
  int kh = rr / 3, kw = rr % 3;
  dst[i] = (bf16)W[(((size_t)ci * Cout + co) * 3 + (2 - kh)) * 3 + (2 - kw)];
}

// ======================= LDS-staged, software-pipelined GEMM =========================
// C[MxN] = A[MxK] * B[NxK]^T, all bf16, K multiple of 32 (rows zero-padded).
// Block = 8 waves = 128(M) x 64(N) tile. Double-buffered LDS B-tile: step i stores tile
// i+1 (prefetched a full step earlier) into buf[p^1] while waves read tile i from
// buf[p]; one barrier per K-step. A fragments double-buffered in registers. All global
// latency is hidden behind a full step of WMMA + ds work. M-tail: clamped A rows +
// guarded stores (no early return -> barriers block-uniform, EXEC all-1s at WMMAs).
// nchw_hw > 0: C stored NCHW at global position nbase+n with per-channel bias + act.
// Else row-major (ldc) with per-column bias + act and optional bf16 dual store.
__global__ void gemm_wmma_kernel(const bf16* __restrict__ A, int lda,
                                 const bf16* __restrict__ Bw,
                                 const float* __restrict__ bias,
                                 float* __restrict__ C, int ldc,
                                 bf16* __restrict__ Cb, int ldcb,
                                 int M, int N, int K, int act,
                                 int nchw_hw, int nbase) {
  __shared__ bf16 lb[2][64 * LDS_STRIDE];
  const int tid = threadIdx.x;
  const int wave = tid >> 5, lane = tid & 31;
  const int lrow = lane & 15, half = lane >> 4;
  const int ngrp = (N + 63) >> 6;
  const int mtb = blockIdx.x / ngrp, ng = blockIdx.x % ngrp;
  const int mt = mtb * 8 + wave;
  const int m = mt * 16 + lrow;
  const int mc = m < M ? m : M - 1;
  const bf16* __restrict__ Arow = A + (size_t)mc * lda;

  // cooperative B staging: thread t -> column nl = t>>2, k-chunk = (t&3)*8
  const int nl = tid >> 2;
  const int chunk = (tid & 3) * 8;
  const int ngl = ng * 64 + nl;
  const int ncl = ngl < N ? ngl : N - 1;
  const bf16* __restrict__ BrowT = Bw + (size_t)ncl * K + chunk;
  const int stoff = nl * LDS_STRIDE + chunk;
  const int rdoff = lrow * LDS_STRIDE + 8 * half;

  int nsub[4];
#pragma unroll
  for (int j = 0; j < 4; ++j) nsub[j] = ng * 64 + j * 16 + lrow;

  const int kcl = K - 32;                      // clamp for harmless over-prefetch
  // ---- prologue: tile 0 staged, tile 1 prefetched ----
  bf16x8 tB = *(const bf16x8*)(BrowT);
  *(bf16x8*)(&lb[0][0] + stoff) = tB;
  int k1 = 32 <= kcl ? 32 : kcl;
  tB = *(const bf16x8*)(BrowT + k1);
  V16 faC, faN;
  faC.h[0] = *(const bf16x8*)(Arow + 8 * half);
  faC.h[1] = *(const bf16x8*)(Arow + 16 + 8 * half);
  faN.h[0] = *(const bf16x8*)(Arow + k1 + 8 * half);
  faN.h[1] = *(const bf16x8*)(Arow + k1 + 16 + 8 * half);
  __syncthreads();

  v8f acc0 = {}, acc1 = {}, acc2 = {}, acc3 = {};
  int p = 0;
  for (int k0 = 0; k0 < K; k0 += 32) {
    // stage tile i+1 (prefetched last step) into the other buffer
    *(bf16x8*)(&lb[p ^ 1][0] + stoff) = tB;
    // prefetch tile i+2 (B) and its A fragment
    int k2 = k0 + 64 <= kcl ? k0 + 64 : kcl;
    tB = *(const bf16x8*)(BrowT + k2);
    V16 faNew;
    faNew.h[0] = *(const bf16x8*)(Arow + k2 + 8 * half);
    faNew.h[1] = *(const bf16x8*)(Arow + k2 + 16 + 8 * half);
    // read tile i fragments from current buffer
    const bf16* __restrict__ lbp = &lb[p][0] + rdoff;
    V16 fb0, fb1, fb2, fb3;
    fb0.h[0] = *(const bf16x8*)(lbp);
    fb0.h[1] = *(const bf16x8*)(lbp + 16);
    fb1.h[0] = *(const bf16x8*)(lbp + 16 * LDS_STRIDE);
    fb1.h[1] = *(const bf16x8*)(lbp + 16 * LDS_STRIDE + 16);
    fb2.h[0] = *(const bf16x8*)(lbp + 32 * LDS_STRIDE);
    fb2.h[1] = *(const bf16x8*)(lbp + 32 * LDS_STRIDE + 16);
    fb3.h[0] = *(const bf16x8*)(lbp + 48 * LDS_STRIDE);
    fb3.h[1] = *(const bf16x8*)(lbp + 48 * LDS_STRIDE + 16);
    acc0 = __builtin_amdgcn_wmma_f32_16x16x32_bf16(false, faC.v, false, fb0.v, (short)0, acc0, false, false);
    acc1 = __builtin_amdgcn_wmma_f32_16x16x32_bf16(false, faC.v, false, fb1.v, (short)0, acc1, false, false);
    acc2 = __builtin_amdgcn_wmma_f32_16x16x32_bf16(false, faC.v, false, fb2.v, (short)0, acc2, false, false);
    acc3 = __builtin_amdgcn_wmma_f32_16x16x32_bf16(false, faC.v, false, fb3.v, (short)0, acc3, false, false);
    faC = faN;
    faN = faNew;
    __syncthreads();     // stores of tile i+1 visible; reads of tile i complete
    p ^= 1;
  }

#pragma unroll
  for (int v = 0; v < 8; ++v) {
    int mo = mt * 16 + v + 8 * half;
    if (mo < M) {
      float a0 = acc0[v], a1 = acc1[v], a2 = acc2[v], a3 = acc3[v];
      if (nchw_hw > 0) {
        int hw = nchw_hw;
        float bb = bias ? bias[mo] : 0.f;
#pragma unroll
        for (int j = 0; j < 4; ++j) {
          int n = nsub[j];
          float av = j == 0 ? a0 : (j == 1 ? a1 : (j == 2 ? a2 : a3));
          if (n < N) {
            int pg = nbase + n;
            C[((size_t)(pg / hw) * M + mo) * hw + (pg % hw)] = act_f(av + bb, act);
          }
        }
      } else {
        float* Cr = C + (size_t)mo * ldc;
#pragma unroll
        for (int j = 0; j < 4; ++j) {
          int n = nsub[j];
          float av = j == 0 ? a0 : (j == 1 ? a1 : (j == 2 ? a2 : a3));
          if (n < N) {
            float val = act_f(av + (bias ? bias[n] : 0.f), act);
            Cr[n] = val;
            if (Cb) Cb[(size_t)mo * ldcb + n] = (bf16)val;
          }
        }
      }
    }
  }
}

// ======================= im2col (bf16 -> bf16), K-padded rows ========================
// MODE: 0 = zero pad, 1 = reflect pad, 2 = transpose-conv (x2 dilated input, k=3).
// col[nl*Kp + k] for global output positions nbase+nl. k >= KK written as zero.
template <int KH, int KW, int MODE>
__global__ void im2col_kernel(const bf16* __restrict__ X, bf16* __restrict__ col,
                              int Cin, int Hin, int Win, int Hout, int Wout,
                              int stride, int pad, int KK, int Kp, int nbase,
                              long long total) {
  long long idx = (long long)blockIdx.x * blockDim.x + threadIdx.x;
  if (idx >= total) return;
  int k = (int)(idx % Kp);
  int n = nbase + (int)(idx / Kp);
  if (k >= KK) { col[idx] = (bf16)0.f; return; }
  int b = n / (Hout * Wout);
  int r0 = n % (Hout * Wout);
  int oy = r0 / Wout, ox = r0 % Wout;
  int ci = k / (KH * KW);
  int rr = k - ci * (KH * KW);
  int r = rr / KW, s = rr - r * KW;
  const size_t base = ((size_t)b * Cin + ci) * Hin * Win;
  bf16 v;
  if (MODE == 2) {
    int ty = oy + r - 1, tx = ox + s - 1;
    bool ok = (ty >= 0) && (tx >= 0) && !(ty & 1) && !(tx & 1);
    int iy = ty >> 1, ix = tx >> 1;
    ok = ok && (iy < Hin) && (ix < Win);
    iy = iclampi(iy, Hin - 1);
    ix = iclampi(ix, Win - 1);
    bf16 xv = X[base + (size_t)iy * Win + ix];
    v = ok ? xv : (bf16)0.f;
  } else if (MODE == 1) {
    int iy = oy * stride - pad + r;
    int ix = ox * stride - pad + s;
    iy = iy < 0 ? -iy : iy;  iy = iy >= Hin ? 2 * Hin - 2 - iy : iy;
    ix = ix < 0 ? -ix : ix;  ix = ix >= Win ? 2 * Win - 2 - ix : ix;
    v = X[base + (size_t)iy * Win + ix];
  } else {
    int iy = oy * stride - pad + r;
    int ix = ox * stride - pad + s;
    bool ok = (iy >= 0) && (iy < Hin) && (ix >= 0) && (ix < Win);
    iy = iclampi(iy, Hin - 1);
    ix = iclampi(ix, Win - 1);
    bf16 xv = X[base + (size_t)iy * Win + ix];
    v = ok ? xv : (bf16)0.f;
  }
  col[idx] = v;
}

// ---------------- instance norm (in-place) + optional bf16 mirror ----------------
__global__ void instnorm_kernel(float* __restrict__ X, bf16* __restrict__ Xb, int HW, int relu) {
  __shared__ float s1[256], s2[256];
  int t = threadIdx.x;
  size_t base = (size_t)blockIdx.x * HW;
  float sum = 0.f, sq = 0.f;
  for (int i = t; i < HW; i += 256) { float v = X[base + i]; sum += v; sq += v * v; }
  s1[t] = sum; s2[t] = sq; __syncthreads();
  for (int st = 128; st > 0; st >>= 1) {
    if (t < st) { s1[t] += s1[t + st]; s2[t] += s2[t + st]; }
    __syncthreads();
  }
  float mean = s1[0] / HW;
  float var  = s2[0] / HW - mean * mean;
  float rstd = rsqrtf(fmaxf(var, 0.f) + EPSN);
  for (int i = t; i < HW; i += 256) {
    float v = (X[base + i] - mean) * rstd;
    if (relu) v = fmaxf(v, 0.f);
    X[base + i] = v;
    if (Xb) Xb[base + i] = (bf16)v;
  }
}

// -------- LayerNorm fused with NCHW -> (B, HW, C) transpose; bf16 out; blockDim == C --
__global__ void ln_transpose_kernel(const float* __restrict__ X, const float* __restrict__ g,
                                    const float* __restrict__ b, bf16* __restrict__ seq,
                                    int C, int HW) {
  __shared__ float s1[256], s2[256];
  int c = threadIdx.x;
  int row = blockIdx.x;            // b*HW + l
  int bidx = row / HW, l = row % HW;
  float v = X[((size_t)bidx * C + c) * HW + l];
  s1[c] = v; s2[c] = v * v; __syncthreads();
  for (int st = 128; st > 0; st >>= 1) {
    if (c < st) { s1[c] += s1[c + st]; s2[c] += s2[c + st]; }
    __syncthreads();
  }
  float mean = s1[0] / C;
  float var  = s2[0] / C - mean * mean;
  float rstd = rsqrtf(fmaxf(var, 0.f) + EPSN);
  seq[(size_t)row * C + c] = (bf16)((v - mean) * rstd * g[c] + b[c]);
}

// ---------------- causal depthwise conv1d (dconv=4) + bias + SiLU, f32+bf16 out ------
__global__ void dwconv_silu_kernel(const float* __restrict__ xz, int ldin,
                                   const float* __restrict__ w, const float* __restrict__ bias,
                                   float* __restrict__ xc, bf16* __restrict__ xcb,
                                   int L, int di, long long total) {
  long long i = (long long)blockIdx.x * blockDim.x + threadIdx.x;
  if (i >= total) return;
  int c = (int)(i % di);
  long long bl = i / di;
  int l = (int)(bl % L);
  long long b = bl / L;
  float s = 0.f;
#pragma unroll
  for (int j = 0; j < 4; ++j) {
    int ll = l - 3 + j;
    if (ll >= 0) s += xz[(size_t)(b * L + ll) * ldin + c] * w[c * 4 + j];
  }
  s += bias[c];
  float r = s / (1.f + __expf(-s));
  xc[i] = r;
  xcb[i] = (bf16)r;
}

// ---------------- selective scan: grid = B, blockDim = di(512), ds = 16 --------------
__global__ void scan_kernel(const float* __restrict__ delta, const float* __restrict__ xdbl,
                            const float* __restrict__ xc, const float* __restrict__ A_log,
                            float* __restrict__ ys, int L, int di) {
  __shared__ float BC[32];
  int d = threadIdx.x;
  int b = blockIdx.x;
  float h[16], Ad[16];
#pragma unroll
  for (int s = 0; s < 16; ++s) { h[s] = 0.f; Ad[s] = -__expf(A_log[d * 16 + s]); }
  for (int l = 0; l < L; ++l) {
    size_t row = (size_t)b * L + l;
    if (d < 32) BC[d] = xdbl[row * 48 + 16 + d];   // [0..15]=B, [16..31]=C
    __syncthreads();
    float dt = delta[row * di + d];
    float xv = xc[row * di + d];
    float dx = dt * xv;
    float y = 0.f;
#pragma unroll
    for (int s = 0; s < 16; ++s) {
      h[s] = h[s] * __expf(dt * Ad[s]) + dx * BC[s];
      y += h[s] * BC[16 + s];
    }
    ys[row * di + d] = y;
    __syncthreads();
  }
}

// ---------------- y = (ys + xc*D) * silu(z), bf16 out --------------------------------
__global__ void ymix_kernel(const float* __restrict__ ys, const float* __restrict__ xc,
                            const float* __restrict__ z, int ldz,
                            const float* __restrict__ D, bf16* __restrict__ y,
                            int di, long long total) {
  long long i = (long long)blockIdx.x * blockDim.x + threadIdx.x;
  if (i >= total) return;
  int c = (int)(i % di);
  long long row = i / di;
  float zz = z[(size_t)row * ldz + c];
  float sz = zz / (1.f + __expf(-zz));
  y[i] = (bf16)((ys[i] + xc[i] * D[c]) * sz);
}

// ---------------- concat [x (NCHW) ; moseq (B,HW,C)] -> bf16 (B, 2C, HW) -------------
__global__ void concat_kernel(const float* __restrict__ x, const float* __restrict__ moseq,
                              bf16* __restrict__ out, int C, int HW, long long total) {
  long long i = (long long)blockIdx.x * blockDim.x + threadIdx.x;
  if (i >= total) return;
  int p = (int)(i % HW);
  long long r = i / HW;
  int c2 = (int)(r % (2 * C));
  long long b = r / (2 * C);
  float v = (c2 < C) ? x[((size_t)b * C + c2) * HW + p]
                     : moseq[((size_t)b * HW + p) * C + (c2 - C)];
  out[i] = (bf16)v;
}

__global__ void add_kernel(const float* __restrict__ a, const float* __restrict__ b,
                           float* __restrict__ o, bf16* __restrict__ ob,
                           long long n, int relu) {
  long long i = (long long)blockIdx.x * blockDim.x + threadIdx.x;
  if (i >= n) return;
  float v = a[i] + b[i];
  if (relu) v = fmaxf(v, 0.f);
  o[i] = v;
  if (ob) ob[i] = (bf16)v;
}

// =====================================================================================
extern "C" void kernel_launch(void* const* d_in, const int* in_sizes, int n_in,
                              void* d_out, int out_size, void* d_ws, size_t ws_size,
                              hipStream_t stream) {
  (void)in_sizes; (void)n_in; (void)out_size; (void)ws_size;
  auto F = [&](int i) { return (const float*)d_in[i]; };
  const int Bn = 2, L = 4096, HWc = 4096;
  const int SLICE = 8192;   // output positions per im2col+GEMM slice

  // ---- workspace bump allocator ----
  char* wsb = (char*)d_ws;
  size_t off = 0;
  auto alloc_f = [&](size_t n) -> float* {
    float* p = (float*)(wsb + off);
    off += ((n * 4 + 255) / 256) * 256;
    return p;
  };
  auto alloc_h = [&](size_t n) -> bf16* {
    bf16* p = (bf16*)(wsb + off);
    off += ((n * 2 + 255) / 256) * 256;
    return p;
  };

  // f32 feature buffers
  float* fA   = alloc_f((size_t)Bn * 64  * 256 * 256);
  float* fB   = alloc_f((size_t)Bn * 128 * 128 * 128);
  float* fC   = alloc_f((size_t)Bn * 256 * 64 * 64);
  float* fC2  = alloc_f((size_t)Bn * 256 * 64 * 64);
  float* fC3  = alloc_f((size_t)Bn * 256 * 64 * 64);
  float* xz   = alloc_f((size_t)Bn * L * 1024);
  float* xcb  = alloc_f((size_t)Bn * L * 512);
  float* xdbl = alloc_f((size_t)Bn * L * 48);
  float* delt = alloc_f((size_t)Bn * L * 512);
  float* ysb  = alloc_f((size_t)Bn * L * 512);
  float* mosq = alloc_f((size_t)Bn * L * 256);
  // bf16 mirrors / operands
  bf16* x_bf    = alloc_h((size_t)Bn * 3 * 256 * 256);
  bf16* fA_bf   = alloc_h((size_t)Bn * 64  * 256 * 256);
  bf16* fB_bf   = alloc_h((size_t)Bn * 128 * 128 * 128);
  bf16* fC_bf   = alloc_h((size_t)Bn * 256 * 64 * 64);
  bf16* fC2_bf  = alloc_h((size_t)Bn * 256 * 64 * 64);
  bf16* cat_bf  = alloc_h((size_t)Bn * 512 * 64 * 64);
  bf16* seq_bf  = alloc_h((size_t)Bn * L * 256);
  bf16* xcb_bf  = alloc_h((size_t)Bn * L * 512);
  bf16* xdbl_bf = alloc_h((size_t)Bn * L * 64);
  bf16* ybuf_bf = alloc_h((size_t)Bn * L * 512);
  bf16* colbuf  = alloc_h((size_t)SLICE * 4608);      // im2col slice scratch (max Kp)

  // packed weights (row stride = Kp, zero-padded)
  bf16* w_stem  = alloc_h((size_t)64 * 160);          // K=147 -> 160
  bf16* w_down0 = alloc_h((size_t)128 * 576);
  bf16* w_down1 = alloc_h((size_t)256 * 1152);
  bf16* w_head  = alloc_h((size_t)3 * 3136);
  bf16* w_up0   = alloc_h((size_t)128 * 2304);
  bf16* w_up1   = alloc_h((size_t)64 * 1152);
  bf16 *w_res1[9], *w_res2[9];
  for (int i = 0; i < 9; ++i) { w_res1[i] = alloc_h((size_t)256 * 2304); w_res2[i] = alloc_h((size_t)256 * 2304); }
  bf16 *w_inp[4], *w_xp[4], *w_dtp[4], *w_outp[4], *w_cc1[4], *w_cc2[4], *w_ccs[4];
  for (int i = 0; i < 4; ++i) {
    w_inp[i]  = alloc_h((size_t)1024 * 256);
    w_xp[i]   = alloc_h((size_t)48 * 512);
    w_dtp[i]  = alloc_h((size_t)512 * 32);            // K=16 -> 32
    w_outp[i] = alloc_h((size_t)256 * 512);
    w_cc1[i]  = alloc_h((size_t)256 * 4608);
    w_cc2[i]  = alloc_h((size_t)256 * 2304);
    w_ccs[i]  = alloc_h((size_t)256 * 512);
  }

  auto packw = [&](const float* s, bf16* d, int rows, int K, int Kp) {
    long long total = (long long)rows * Kp;
    pack_wpad_kernel<<<dim3((int)((total + 255) / 256)), dim3(256), 0, stream>>>(s, d, K, Kp, total);
  };
  // Input map (pytree dict order): 0:down0.b 1:down0.w 2:down1.b 3:down1.w 4:head_b
  // 5:head_w  6+14*i: mamba[i]{A_log,D,cc_skip_w,cc_w1,cc_w2,conv_b,conv_w,dt_proj_b,
  // dt_proj_w,in_proj_w,ln_b,ln_g,out_proj_w,x_proj_w}  62+4*i: res[i]{b1,b2,w1,w2}
  // 98:stem_b 99:stem_w 100..103:up 104:x
  packw(F(99), w_stem, 64, 147, 160);
  packw(F(1),  w_down0, 128, 576, 576);
  packw(F(3),  w_down1, 256, 1152, 1152);
  packw(F(5),  w_head, 3, 3136, 3136);
  pack_convT_kernel<<<dim3((256 * 128 * 9 + 255) / 256), dim3(256), 0, stream>>>(F(101), w_up0, 256, 128);
  pack_convT_kernel<<<dim3((128 * 64 * 9 + 255) / 256),  dim3(256), 0, stream>>>(F(103), w_up1, 128, 64);
  for (int i = 0; i < 9; ++i) {
    int b0 = 62 + 4 * i;
    packw(F(b0 + 2), w_res1[i], 256, 2304, 2304);
    packw(F(b0 + 3), w_res2[i], 256, 2304, 2304);
  }
  for (int i = 0; i < 4; ++i) {
    int b0 = 6 + 14 * i;
    packw(F(b0 + 2),  w_ccs[i],  256, 512, 512);
    packw(F(b0 + 3),  w_cc1[i],  256, 4608, 4608);
    packw(F(b0 + 4),  w_cc2[i],  256, 2304, 2304);
    packw(F(b0 + 8),  w_dtp[i],  512, 16, 32);
    packw(F(b0 + 9),  w_inp[i],  1024, 256, 256);
    packw(F(b0 + 12), w_outp[i], 256, 512, 512);
    packw(F(b0 + 13), w_xp[i],   48, 512, 512);
  }
  pack_bf16_kernel<<<dim3((Bn * 3 * 65536 + 255) / 256), dim3(256), 0, stream>>>(F(104), x_bf, Bn * 3 * 65536);

  auto gemm = [&](const bf16* Ap, int lda, const bf16* Bp, const float* bias, float* Cp,
                  int ldc, bf16* Cbp, int ldcb, int M, int N, int K, int act, int hw, int nb) {
    int mtblocks = ((M + 15) / 16 + 7) / 8;
    int blocks = mtblocks * ((N + 63) / 64);
    gemm_wmma_kernel<<<dim3(blocks), dim3(256), 0, stream>>>(Ap, lda, Bp, bias, Cp, ldc,
                                                             Cbp, ldcb, M, N, K, act, hw, nb);
  };
  // conv via sliced im2col + LDS GEMM (NCHW store)
  auto convg = [&](const bf16* Xbp, const bf16* Wp, const float* bias, float* Yp,
                   int Ci, int Hi, int Wi, int Co, int Ho, int Wo,
                   int kh2, int st, int pd, int mode, int act, int KK, int Kp) {
    int Np = Bn * Ho * Wo;
    for (int nb = 0; nb < Np; nb += SLICE) {
      int ns = Np - nb < SLICE ? Np - nb : SLICE;
      long long total = (long long)ns * Kp;
      int blocks = (int)((total + 255) / 256);
      if (kh2 == 7)
        im2col_kernel<7, 7, 1><<<dim3(blocks), dim3(256), 0, stream>>>(
            Xbp, colbuf, Ci, Hi, Wi, Ho, Wo, st, pd, KK, Kp, nb, total);
      else if (mode == 2)
        im2col_kernel<3, 3, 2><<<dim3(blocks), dim3(256), 0, stream>>>(
            Xbp, colbuf, Ci, Hi, Wi, Ho, Wo, st, pd, KK, Kp, nb, total);
      else if (mode == 1)
        im2col_kernel<3, 3, 1><<<dim3(blocks), dim3(256), 0, stream>>>(
            Xbp, colbuf, Ci, Hi, Wi, Ho, Wo, st, pd, KK, Kp, nb, total);
      else if (kh2 == 1)
        im2col_kernel<1, 1, 0><<<dim3(blocks), dim3(256), 0, stream>>>(
            Xbp, colbuf, Ci, Hi, Wi, Ho, Wo, st, pd, KK, Kp, nb, total);
      else
        im2col_kernel<3, 3, 0><<<dim3(blocks), dim3(256), 0, stream>>>(
            Xbp, colbuf, Ci, Hi, Wi, Ho, Wo, st, pd, KK, Kp, nb, total);
      gemm(Wp, Kp, colbuf, bias, Yp, 0, nullptr, 0, Co, ns, Kp, act, Ho * Wo, nb);
    }
  };
  auto inorm = [&](float* Xp, bf16* Xbp, int planes, int HW, int relu) {
    instnorm_kernel<<<dim3(planes), dim3(256), 0, stream>>>(Xp, Xbp, HW, relu);
  };
  auto addk = [&](const float* a, const float* b2, float* o, bf16* ob, long long n, int relu) {
    add_kernel<<<dim3((int)((n + 255) / 256)), dim3(256), 0, stream>>>(a, b2, o, ob, n, relu);
  };

  auto cm_mamba = [&](int mi) {
    int b0 = 6 + 14 * mi;
    const float* A_log  = F(b0 + 0);
    const float* Dp     = F(b0 + 1);
    const float* conv_b = F(b0 + 5);
    const float* conv_w = F(b0 + 6);
    const float* dt_b   = F(b0 + 7);
    const float* ln_b   = F(b0 + 10);
    const float* ln_g   = F(b0 + 11);
    ln_transpose_kernel<<<dim3(Bn * HWc), dim3(256), 0, stream>>>(fC, ln_g, ln_b, seq_bf, 256, HWc);
    gemm(seq_bf, 256, w_inp[mi], nullptr, xz, 1024, nullptr, 0, Bn * L, 1024, 256, 0, 0, 0);
    {
      long long total = (long long)Bn * L * 512;
      dwconv_silu_kernel<<<dim3((int)((total + 255) / 256)), dim3(256), 0, stream>>>(
          xz, 1024, conv_w, conv_b, xcb, xcb_bf, L, 512, total);
    }
    gemm(xcb_bf, 512, w_xp[mi], nullptr, xdbl, 48, xdbl_bf, 64, Bn * L, 48, 512, 0, 0, 0);
    gemm(xdbl_bf, 64, w_dtp[mi], dt_b, delt, 512, nullptr, 0, Bn * L, 512, 32, 2, 0, 0);  // softplus
    scan_kernel<<<dim3(Bn), dim3(512), 0, stream>>>(delt, xdbl, xcb, A_log, ysb, L, 512);
    {
      long long total = (long long)Bn * L * 512;
      ymix_kernel<<<dim3((int)((total + 255) / 256)), dim3(256), 0, stream>>>(
          ysb, xcb, xz + 512, 1024, Dp, ybuf_bf, 512, total);
    }
    gemm(ybuf_bf, 512, w_outp[mi], nullptr, mosq, 256, nullptr, 0, Bn * L, 256, 512, 0, 0, 0);
    {
      long long total = (long long)Bn * 512 * HWc;
      concat_kernel<<<dim3((int)((total + 255) / 256)), dim3(256), 0, stream>>>(
          fC, mosq, cat_bf, 256, HWc, total);
    }
    convg(cat_bf, w_cc1[mi], nullptr, fC2, 512, 64, 64, 256, 64, 64, 3, 1, 1, 0, 0, 4608, 4608);
    inorm(fC2, fC2_bf, Bn * 256, HWc, 1);
    convg(fC2_bf, w_cc2[mi], nullptr, fC3, 256, 64, 64, 256, 64, 64, 3, 1, 1, 0, 0, 2304, 2304);
    inorm(fC3, nullptr, Bn * 256, HWc, 0);
    convg(cat_bf, w_ccs[mi], nullptr, fC2, 512, 64, 64, 256, 64, 64, 1, 1, 0, 0, 0, 512, 512);
    inorm(fC2, nullptr, Bn * 256, HWc, 0);
    addk(fC3, fC2, fC, fC_bf, (long long)Bn * 256 * HWc, 1);
  };

  // ================= pipeline =================
  convg(x_bf, w_stem, nullptr, fA, 3, 256, 256, 64, 256, 256, 7, 1, 3, 1, 0, 147, 160);
  inorm(fA, fA_bf, Bn * 64, 65536, 1);
  convg(fA_bf, w_down0, nullptr, fB, 64, 256, 256, 128, 128, 128, 3, 2, 1, 0, 0, 576, 576);
  inorm(fB, fB_bf, Bn * 128, 16384, 1);
  convg(fB_bf, w_down1, nullptr, fC, 128, 128, 128, 256, 64, 64, 3, 2, 1, 0, 0, 1152, 1152);
  inorm(fC, fC_bf, Bn * 256, 4096, 1);

  cm_mamba(0);
  int mi = 1;
  for (int i = 0; i < 9; ++i) {
    convg(fC_bf, w_res1[i], nullptr, fC2, 256, 64, 64, 256, 64, 64, 3, 1, 1, 1, 0, 2304, 2304);
    inorm(fC2, fC2_bf, Bn * 256, 4096, 1);
    convg(fC2_bf, w_res2[i], nullptr, fC3, 256, 64, 64, 256, 64, 64, 3, 1, 1, 1, 0, 2304, 2304);
    inorm(fC3, nullptr, Bn * 256, 4096, 0);
    addk(fC, fC3, fC, fC_bf, (long long)Bn * 256 * 4096, 0);
    if ((i + 1) % 3 == 0 || i == 8) { cm_mamba(mi); ++mi; }
  }

  convg(fC_bf, w_up0, nullptr, fB, 256, 64, 64, 128, 128, 128, 3, 1, 0, 2, 0, 2304, 2304);   // convT
  inorm(fB, fB_bf, Bn * 128, 16384, 1);
  convg(fB_bf, w_up1, nullptr, fA, 128, 128, 128, 64, 256, 256, 3, 1, 0, 2, 0, 1152, 1152);  // convT
  inorm(fA, fA_bf, Bn * 64, 65536, 1);
  convg(fA_bf, w_head, F(4), (float*)d_out, 64, 256, 256, 3, 256, 256, 7, 1, 3, 1, 3, 3136, 3136); // tanh
}
